// DualTransformerLayerOLD_23416161698001
// MI455X (gfx1250) — matrix operand order repaired and checked by reading, not compiled, not run on hardware
//
#include <hip/hip_runtime.h>
#include <math.h>

// Problem dims (match reference)
#define Bn  32
#define Nn  400
#define Hn  512
#define NHn 8
#define Pn  3
#define DKn 64

static const long long BNH  = (long long)Bn * Nn * Hn;        // 6,553,600
static const long long BN3H = (long long)Bn * Nn * 3 * Hn;    // 19,660,800
static const long long BNN  = (long long)Bn * Nn * Nn;        // 5,120,000
static const long long BN4H = (long long)Bn * Nn * 4 * Hn;    // 26,214,400

typedef __attribute__((ext_vector_type(16))) _Float16 v16h;
typedef __attribute__((ext_vector_type(8)))  _Float16 v8h;
typedef __attribute__((ext_vector_type(4)))  _Float16 v4h;
typedef __attribute__((ext_vector_type(8)))  float    v8f;

__device__ inline v8f zero_v8f() {
    v8f z;
#pragma unroll
    for (int i = 0; i < 8; ++i) z[i] = 0.0f;
    return z;
}

// 8 contiguous floats -> 8 f16 (two 16B vector loads; caller guarantees 16B align)
__device__ inline v8h cvt8(const float* p) {
    const float4 a = *(const float4*)p;
    const float4 b = *(const float4*)(p + 4);
    v8h r;
    r[0] = (_Float16)a.x; r[1] = (_Float16)a.y; r[2] = (_Float16)a.z; r[3] = (_Float16)a.w;
    r[4] = (_Float16)b.x; r[5] = (_Float16)b.y; r[6] = (_Float16)b.z; r[7] = (_Float16)b.w;
    return r;
}

__device__ inline v4h cvt4(float4 a) {
    v4h r;
    r[0] = (_Float16)a.x; r[1] = (_Float16)a.y; r[2] = (_Float16)a.z; r[3] = (_Float16)a.w;
    return r;
}

__device__ inline v16h combine(v8h lo, v8h hi) {
    return __builtin_shufflevector(lo, hi, 0, 1, 2, 3, 4, 5, 6, 7,
                                          8, 9, 10, 11, 12, 13, 14, 15);
}

// ---------------------------------------------------------------------------
// Batched GEMM: C[b] = act(A[b] @ B[b] + bias). 128 threads = 4 waves.
// Block tile 128x64 (wave tile 32x64 -> 8 WMMAs per 32-K chunk per wave).
// A/B staged as f16 in LDS; B transposed so fragments are contiguous 16B loads.
// Requirements (all call sites obey): M % 16 == 0, Nc % 64 == 0. K arbitrary
// (tail chunk uses clamped-address + select-zero, fully branchless).
// act: 0 = none, 1 = exact GELU.
// ---------------------------------------------------------------------------
__global__ void gemm_wmma_kernel(const float* __restrict__ A,
                                 const float* __restrict__ Bm,
                                 const float* __restrict__ bias,
                                 float* __restrict__ C,
                                 int M, int Nc, int K,
                                 int lda, int ldb, int ldc,
                                 long long sA, long long sB, long long sC,
                                 int act) {
    // row stride 40 f16 = 80B keeps every 8-f16 run 16B aligned (and 4-f16 run 8B)
    __shared__ __align__(16) _Float16 As[128][40];  // As[row][k]
    __shared__ __align__(16) _Float16 Bs[64][40];   // Bs[col][k] (transposed)

    const int bz = blockIdx.z;
    A  += (long long)bz * sA;
    Bm += (long long)bz * sB;
    C  += (long long)bz * sC;

    const int tid  = threadIdx.x;
    const int wave = tid >> 5;
    const int lane = tid & 31;
    const int rowb = blockIdx.y * 128;
    const int colb = blockIdx.x * 64;
    const bool rowfull = (rowb + 128 <= M);

    const int khalf = (lane & 16) ? 8 : 0;    // unused; A frags also use kbase
    const int kbase = (lane & 16) ? 16 : 0;   // B-frag K sub-offset
    const int cl    = lane & 15;
    const int rl0   = wave * 32 + cl;         // LDS row of m-frag 0
    const int rl1   = rl0 + 16;               // LDS row of m-frag 1

    v8f acc[2][4] = { { zero_v8f(), zero_v8f(), zero_v8f(), zero_v8f() },
                      { zero_v8f(), zero_v8f(), zero_v8f(), zero_v8f() } };

    for (int kk = 0; kk < K; kk += 32) {
        const bool full = rowfull && (kk + 32 <= K);   // block-uniform
        if (full) {
            // ---- A: 128 rows x 32 k = 1024 float4s, 8 per thread ----
#pragma unroll
            for (int q = 0; q < 8; ++q) {
                const int f  = q * 128 + tid;
                const int r  = f >> 3;            // 8 float4 per row
                const int kq = (f & 7) * 4;
                const float* ap = A + (long long)(rowb + r) * lda + kk + kq;
                __builtin_prefetch(ap + 32, 0, 1);
                *(v4h*)&As[r][kq] = cvt4(*(const float4*)ap);
            }
            // ---- B: 32 k x 64 cols = 512 float4s, 4 per thread, transposed ----
#pragma unroll
            for (int q = 0; q < 4; ++q) {
                const int f  = q * 128 + tid;
                const int kr = f >> 4;            // 16 float4 per k-row
                const int cq = (f & 15) * 4;
                const float4 v = *(const float4*)(Bm + (long long)(kk + kr) * ldb + colb + cq);
                Bs[cq + 0][kr] = (_Float16)v.x;
                Bs[cq + 1][kr] = (_Float16)v.y;
                Bs[cq + 2][kr] = (_Float16)v.z;
                Bs[cq + 3][kr] = (_Float16)v.w;
            }
        } else {
            // Tail / partial-row chunk: clamped addresses + select-zero.
            // Statically unrolled so there is no divergent loop guard.
#pragma unroll
            for (int q = 0; q < 32; ++q) {        // 4096 A elements
                const int e  = q * 128 + tid;
                const int r  = e >> 5, kc = e & 31;
                const int rg  = rowb + r;
                const int rcl = rg < M ? rg : (M - 1);
                const int kg  = kk + kc;
                const int kcl = kg < K ? kg : (K - 1);
                const float v = A[(long long)rcl * lda + kcl];
                As[r][kc] = (_Float16)(kg < K ? v : 0.0f);
            }
#pragma unroll
            for (int q = 0; q < 16; ++q) {        // 2048 B elements
                const int e  = q * 128 + tid;
                const int kr = e >> 6, c = e & 63;
                const int kg  = kk + kr;
                const int kcl = kg < K ? kg : (K - 1);
                const float v = Bm[(long long)kcl * ldb + colb + c];
                Bs[c][kr] = (_Float16)(kg < K ? v : 0.0f);
            }
        }
        __syncthreads();

        // ---- fragments from LDS (aligned 16B loads) + 8 WMMAs ----
        const v16h a0 = combine(*(const v8h*)&As[rl0][kbase],
                                *(const v8h*)&As[rl0][kbase + 8]);
        const v16h a1 = combine(*(const v8h*)&As[rl1][kbase],
                                *(const v8h*)&As[rl1][kbase + 8]);
#pragma unroll
        for (int t = 0; t < 4; ++t) {
            const _Float16* bp = &Bs[t * 16 + cl][kbase];
            const v16h b = combine(*(const v8h*)bp, *(const v8h*)(bp + 8));
            acc[0][t] = __builtin_amdgcn_wmma_f32_16x16x32_f16(false, a0, false, b, (short)0, acc[0][t], false, false);
            acc[1][t] = __builtin_amdgcn_wmma_f32_16x16x32_f16(false, a1, false, b, (short)0, acc[1][t], false, false);
        }
        __syncthreads();
    }

    // ---- epilogue: M%16==0 & Nc%64==0 -> wave-uniform row guards only ----
    const int mo = (lane & 16) ? 8 : 0;
#pragma unroll
    for (int mi = 0; mi < 2; ++mi) {
        const int r0 = rowb + wave * 32 + mi * 16;
        if (r0 < M) {
#pragma unroll
            for (int t = 0; t < 4; ++t) {
                const int col = colb + t * 16 + cl;
                const float bv = bias ? bias[col] : 0.0f;
#pragma unroll
                for (int j = 0; j < 8; ++j) {
                    const int r = r0 + mo + j;
                    float v = acc[mi][t][j] + bv;
                    if (act == 1) v = 0.5f * v * (1.0f + erff(v * 0.70710678118654752f));
                    C[(long long)r * ldc + col] = v;
                }
            }
        }
    }
}

// ---------------------------------------------------------------------------
// Fused attention: block = (b, h, 16-query tile), 128 threads = 4 waves.
// Phase 1: S = (Q Kt)*scale*(1+mask) -> LDS (16 x 416, cols 400..415 zero).
// Phase 2: parallel row softmax (16 rows x 8 segments).
// Phase 3: ctx = P @ V via WMMA, P fragments vector-loaded from LDS.
// ---------------------------------------------------------------------------
__global__ void attn_kernel(const float* __restrict__ qkv,
                            const float* __restrict__ wmask,
                            float* __restrict__ ctx) {
    const int qt = blockIdx.x;   // 0..24
    const int h  = blockIdx.y;   // 0..7
    const int b  = blockIdx.z;   // 0..31
    const int tid  = threadIdx.x;
    const int wave = tid >> 5;
    const int lane = tid & 31;

    __shared__ __align__(16) float s[16][416];
    __shared__ float rred[16][9];
    for (int idx = tid; idx < 16 * 16; idx += 128)
        s[idx >> 4][400 + (idx & 15)] = 0.0f;

    const long long rs = 3LL * Hn;
    const float* qbase = qkv + ((long long)b * Nn) * rs + h * DKn;
    const float* kbase = qbase + Hn;
    const float* vbase = qbase + 2 * Hn;

    const int khalf = (lane & 16) ? 8 : 0;
    const int kb16  = (lane & 16) ? 16 : 0;
    const int cl    = lane & 15;
    const int qrow  = qt * 16 + cl;           // < 400 always

    // Q fragments: contiguous float4 runs
    const float* qp = qbase + (long long)qrow * rs;
    const v16h aq0 = combine(cvt8(qp + kb16),      cvt8(qp + kb16 + 8));
    const v16h aq1 = combine(cvt8(qp + 32 + kb16), cvt8(qp + 32 + kb16 + 8));

    // ---- Phase 1: scores ----
    const float scale = 0.125f;  // 64^-0.5
    for (int j = wave; j < 25; j += 4) {
        v8f acc = zero_v8f();
        const int kcol = j * 16 + cl;                 // key row index
        const float* kp = kbase + (long long)kcol * rs;
        {
            const v16h kb = combine(cvt8(kp + kb16), cvt8(kp + kb16 + 8));
            acc = __builtin_amdgcn_wmma_f32_16x16x32_f16(false, aq0, false, kb, (short)0, acc, false, false);
        }
        {
            const v16h kb = combine(cvt8(kp + 32 + kb16), cvt8(kp + 32 + kb16 + 8));
            acc = __builtin_amdgcn_wmma_f32_16x16x32_f16(false, aq1, false, kb, (short)0, acc, false, false);
        }
        const int mo = (lane & 16) ? 8 : 0;
#pragma unroll
        for (int jj = 0; jj < 8; ++jj) {
            const int m  = jj + mo;
            const int gq = qt * 16 + m;
            const int gk = j * 16 + cl;
            const float mv = wmask[((long long)b * Nn + gq) * Nn + gk];
            s[m][j * 16 + cl] = acc[jj] * scale * (1.0f + mv);
        }
    }
    __syncthreads();

    // ---- Phase 2: softmax, 8 segments of 50 columns per row ----
    const int r2  = tid & 15;
    const int seg = tid >> 4;       // 0..7
    const int cb  = seg * 50;
    {
        float mx = -1e30f;
        for (int c = cb; c < cb + 50; ++c) mx = fmaxf(mx, s[r2][c]);
        rred[r2][seg] = mx;
    }
    __syncthreads();
    if (seg == 0) {
        float m = rred[r2][0];
#pragma unroll
        for (int q = 1; q < 8; ++q) m = fmaxf(m, rred[r2][q]);
        rred[r2][8] = m;
    }
    __syncthreads();
    const float rowm = rred[r2][8];
    {
        float sum = 0.0f;
        for (int c = cb; c < cb + 50; ++c) {
            const float e = expf(s[r2][c] - rowm);
            s[r2][c] = e;
            sum += e;
        }
        rred[r2][seg] = sum;
    }
    __syncthreads();
    if (seg == 0) {
        float t = 0.0f;
#pragma unroll
        for (int q = 0; q < 8; ++q) t += rred[r2][q];
        rred[r2][8] = t;
    }
    __syncthreads();
    {
        const float inv = 1.0f / rred[r2][8];
        for (int c = cb; c < cb + 50; ++c) s[r2][c] *= inv;
    }
    __syncthreads();

    // ---- Phase 3: ctx = P @ V, wave w owns head-cols [w*16, w*16+16) ----
    v8f acc = zero_v8f();
    const int vcol = wave * 16 + cl;
    for (int chunk = 0; chunk < 12; ++chunk) {     // fully in-range chunks
        const float* sp = &s[cl][chunk * 32];
        const v16h pa = combine(cvt8(sp + kb16), cvt8(sp + kb16 + 8));
        v16h vb;
#pragma unroll
        for (int i = 0; i < 16; ++i) {
            const int kr = chunk * 32 + kb16 + i;
            vb[i] = (_Float16)vbase[(long long)kr * rs + vcol];
        }
        acc = __builtin_amdgcn_wmma_f32_16x16x32_f16(false, pa, false, vb, (short)0, acc, false, false);
    }
    {   // tail chunk 12: k rows 384..415, clamp + select-zero (branchless)
        const float* sp = &s[cl][12 * 32];
        const v16h pa = combine(cvt8(sp + kb16), cvt8(sp + kb16 + 8));
        v16h vb;
#pragma unroll
        for (int i = 0; i < 16; ++i) {
            const int kr  = 12 * 32 + kb16 + i;
            const int krc = kr < Nn ? kr : (Nn - 1);
            const float v = vbase[(long long)krc * rs + vcol];
            vb[i] = (_Float16)(kr < Nn ? v : 0.0f);
        }
        acc = __builtin_amdgcn_wmma_f32_16x16x32_f16(false, pa, false, vb, (short)0, acc, false, false);
    }
    const int mo = (lane & 16) ? 8 : 0;
#pragma unroll
    for (int jj = 0; jj < 8; ++jj) {
        const int gq = qt * 16 + jj + mo;
        ctx[((long long)b * Nn + gq) * Hn + h * DKn + wave * 16 + cl] = acc[jj];
    }
}

// ---------------------------------------------------------------------------
// Row LayerNorm over H=512 (256 threads, 2 elements/thread)
// ---------------------------------------------------------------------------
__global__ void ln_kernel(const float* __restrict__ x, const float* __restrict__ g,
                          const float* __restrict__ bt, float* __restrict__ out) {
    const long long row = blockIdx.x;
    const float* xr = x + row * Hn;
    float* orow = out + row * Hn;
    const int tid = threadIdx.x;
    const float v0 = xr[tid], v1 = xr[tid + 256];
    __shared__ float red[256];

    red[tid] = v0 + v1;
    __syncthreads();
    for (int s = 128; s > 0; s >>= 1) {
        if (tid < s) red[tid] += red[tid + s];
        __syncthreads();
    }
    const float mean = red[0] * (1.0f / 512.0f);
    __syncthreads();

    const float d0 = v0 - mean, d1 = v1 - mean;
    red[tid] = d0 * d0 + d1 * d1;
    __syncthreads();
    for (int s = 128; s > 0; s >>= 1) {
        if (tid < s) red[tid] += red[tid + s];
        __syncthreads();
    }
    const float inv = rsqrtf(red[0] * (1.0f / 512.0f) + 1e-5f);
    orow[tid]       = d0 * inv * g[tid]       + bt[tid];
    orow[tid + 256] = d1 * inv * g[tid + 256] + bt[tid + 256];
}

// Row log-softmax over H=512
__global__ void logsoftmax_kernel(const float* __restrict__ x, float* __restrict__ l) {
    const long long row = blockIdx.x;
    const float* xr = x + row * Hn;
    float* lr = l + row * Hn;
    const int tid = threadIdx.x;
    const float v0 = xr[tid], v1 = xr[tid + 256];
    __shared__ float red[256];

    red[tid] = fmaxf(v0, v1);
    __syncthreads();
    for (int s = 128; s > 0; s >>= 1) {
        if (tid < s) red[tid] = fmaxf(red[tid], red[tid + s]);
        __syncthreads();
    }
    const float mx = red[0];
    __syncthreads();
    red[tid] = expf(v0 - mx) + expf(v1 - mx);
    __syncthreads();
    for (int s = 128; s > 0; s >>= 1) {
        if (tid < s) red[tid] += red[tid + s];
        __syncthreads();
    }
    const float lse = mx + logf(red[0]);
    lr[tid]       = v0 - lse;
    lr[tid + 256] = v1 - lse;
}

// zbar[b,h] = mean_n z[b,n,h]
__global__ void colmean_kernel(const float* __restrict__ z, float* __restrict__ zbar) {
    const int idx = blockIdx.x * blockDim.x + threadIdx.x;
    if (idx >= Bn * Hn) return;
    const int b = idx / Hn, hc = idx % Hn;
    const float* p = z + ((long long)b * Nn) * Hn + hc;
    float sum = 0.0f;
    for (int n = 0; n < Nn; ++n) sum += p[(long long)n * Hn];
    zbar[idx] = sum * (1.0f / (float)Nn);
}

// qv[b,j] = zbar[b,:] . wq[:,j]
__global__ void qv_kernel(const float* __restrict__ zbar, const float* __restrict__ wq,
                          float* __restrict__ qv) {
    const int idx = blockIdx.x * blockDim.x + threadIdx.x;
    if (idx >= Bn * Hn) return;
    const int b = idx / Hn, j = idx % Hn;
    float sum = 0.0f;
    for (int k = 0; k < Hn; ++k) sum += zbar[b * Hn + k] * wq[(long long)k * Hn + j];
    qv[idx] = sum;
}

// aw[b,:] = softmax( (qv[b] . keys[p]) * H^-0.5 ) over P=3
__global__ void aw_kernel(const float* __restrict__ qv, const float* __restrict__ keys,
                          float* __restrict__ aw) {
    const int b = blockIdx.x * blockDim.x + threadIdx.x;
    if (b >= Bn) return;
    const float invs = 0.044194173824159216f;  // 512^-0.5
    float lg[Pn];
    float mx = -1e30f;
    for (int p = 0; p < Pn; ++p) {
        float s = 0.0f;
        for (int k = 0; k < Hn; ++k) s += qv[b * Hn + k] * keys[p * Hn + k];
        lg[p] = s * invs;
        mx = fmaxf(mx, lg[p]);
    }
    float sum = 0.0f;
    for (int p = 0; p < Pn; ++p) { lg[p] = expf(lg[p] - mx); sum += lg[p]; }
    const float inv = 1.0f / sum;
    for (int p = 0; p < Pn; ++p) aw[b * Pn + p] = lg[p] * inv;
}

// w[b,i,j] = sum_p aw[b,p] * pri[b,p,i,j]
__global__ void priormix_kernel(const float* __restrict__ aw, const float* __restrict__ pri,
                                float* __restrict__ w, long long total) {
    const long long NN = (long long)Nn * Nn;
    for (long long idx = blockIdx.x * (long long)blockDim.x + threadIdx.x;
         idx < total; idx += (long long)gridDim.x * blockDim.x) {
        const long long b = idx / NN, rem = idx % NN;
        const long long base = b * Pn * NN + rem;
        w[idx] = aw[b * Pn + 0] * pri[base]
               + aw[b * Pn + 1] * pri[base + NN]
               + aw[b * Pn + 2] * pri[base + 2 * NN];
    }
}

__global__ void add3_kernel(float* __restrict__ dst, const float* __restrict__ a,
                            const float* __restrict__ b, long long n) {
    for (long long i = blockIdx.x * (long long)blockDim.x + threadIdx.x;
         i < n; i += (long long)gridDim.x * blockDim.x)
        dst[i] = dst[i] + a[i] + b[i];
}

__global__ void add2_kernel(float* __restrict__ dst, const float* __restrict__ a,
                            const float* __restrict__ b, long long n) {
    for (long long i = blockIdx.x * (long long)blockDim.x + threadIdx.x;
         i < n; i += (long long)gridDim.x * blockDim.x)
        dst[i] = a[i] + b[i];
}

// out = (1-w)*a + w*b, w = dw[sidx]
__global__ void mix_kernel(float* __restrict__ out, const float* __restrict__ a,
                           const float* __restrict__ b, const float* __restrict__ dw,
                           int sidx, long long n) {
    const float w = dw[sidx];
    for (long long i = blockIdx.x * (long long)blockDim.x + threadIdx.x;
         i < n; i += (long long)gridDim.x * blockDim.x)
        out[i] = (1.0f - w) * a[i] + w * b[i];
}

__global__ void zero_kernel(float* __restrict__ p, int n) {
    const int i = blockIdx.x * blockDim.x + threadIdx.x;
    if (i < n) p[i] = 0.0f;
}

// acc[0] += sum p_SC*(l_SC-l_FC); acc[1] += sum p_FC*(l_FC-l_SC)
__global__ void kl_kernel(const float* __restrict__ lF, const float* __restrict__ lS,
                          float* __restrict__ acc, long long total) {
    __shared__ float r0[256], r1[256];
    float s0 = 0.0f, s1 = 0.0f;
    for (long long i = blockIdx.x * (long long)blockDim.x + threadIdx.x;
         i < total; i += (long long)gridDim.x * blockDim.x) {
        const float a = lF[i], c = lS[i];
        s0 += expf(c) * (c - a);
        s1 += expf(a) * (a - c);
    }
    r0[threadIdx.x] = s0; r1[threadIdx.x] = s1;
    __syncthreads();
    for (int s = 128; s > 0; s >>= 1) {
        if (threadIdx.x < s) { r0[threadIdx.x] += r0[threadIdx.x + s];
                               r1[threadIdx.x] += r1[threadIdx.x + s]; }
        __syncthreads();
    }
    if (threadIdx.x == 0) { atomicAdd(&acc[0], r0[0]); atomicAdd(&acc[1], r1[0]); }
}

__global__ void finalize_kernel(const float* __restrict__ acc, const float* __restrict__ dw,
                                float* __restrict__ loss) {
    if (threadIdx.x == 0 && blockIdx.x == 0) {
        const float rows = (float)(Bn * Nn);
        const float klFC = acc[0] / rows;
        const float klSC = acc[1] / rows;
        loss[0] = 0.5f * (klFC * dw[0] + klSC * dw[1]);
    }
}

// ---------------------------------------------------------------------------
extern "C" void kernel_launch(void* const* d_in, const int* in_sizes, int n_in,
                              void* d_out, int out_size, void* d_ws, size_t ws_size,
                              hipStream_t stream) {
    (void)in_sizes; (void)n_in; (void)out_size; (void)ws_size;
    auto in = [&](int i) { return (const float*)d_in[i]; };

    const float* x[2]    = { in(0), in(1) };
    const float* pri[2]  = { in(2), in(3) };
    const float* Cm[2]   = { in(4), in(5) };
    const float* dw      = in(6);
    const float* lnag[2] = { in(7),  in(9)  };
    const float* lnab[2] = { in(8),  in(10) };
    const float* lnfg[2] = { in(11), in(13) };
    const float* lnfb[2] = { in(12), in(14) };
    const float* qkvw[2] = { in(15), in(16) };
    const float* fw1[2]  = { in(17), in(21) };
    const float* fb1[2]  = { in(18), in(22) };
    const float* fw2[2]  = { in(19), in(23) };
    const float* fb2[2]  = { in(20), in(24) };
    const float* wgq   = in(25);
    const float* keys  = in(26);
    const float* wfuse = in(27);
    const float* bfuse = in(28);
    const float* diw[2] = { in(29), in(31) };
    const float* dib[2] = { in(30), in(32) };
    const float* dow[2] = { in(33), in(35) };
    const float* dob[2] = { in(34), in(36) };

    float* out = (float*)d_out;
    float* xout[2]  = { out, out + BNH };
    float* loss     = out + 2 * BNH;
    float* awout[2] = { out + 2 * BNH + 1, out + 2 * BNH + 1 + Bn * Pn };

    // workspace carve-out
    float* W = (float*)d_ws;
    size_t off = 0;
    auto alloc = [&](long long nel) { float* p = W + off; off += (size_t)nel; return p; };
    float* z[2]    = { alloc(BNH),  alloc(BNH)  };
    float* qkv[2]  = { alloc(BN3H), alloc(BN3H) };
    float* wm[2]   = { alloc(BNN),  alloc(BNN)  };
    float* fused[2]= { alloc(BNH),  alloc(BNH)  };
    float* attn[2] = { alloc(BNH),  alloc(BNH)  };
    float* xnew[2] = { alloc(BNH),  alloc(BNH)  };
    float* proj[2] = { alloc(BNH),  alloc(BNH)  };
    float* lsm[2]  = { alloc(BNH),  alloc(BNH)  };
    float* zbar[2] = { alloc(Bn * Hn), alloc(Bn * Hn) };
    float* qv[2]   = { alloc(Bn * Hn), alloc(Bn * Hn) };
    float* h1   = alloc(BN4H);  // shared scratch (streams processed serially)
    float* tbuf = alloc(BNH);
    float* tmp2 = alloc(BNH);
    float* acc  = alloc(8);

    const int M = Bn * Nn;                 // 12800, 100 row-tiles of 128
    const int EB = 16384;                  // elementwise grid (grid-stride loops)

    for (int s = 0; s < 2; ++s) {
        // z = LN(x)
        ln_kernel<<<M, 256, 0, stream>>>(x[s], lnag[s], lnab[s], z[s]);
        // qkv = z @ qkv_w   [12800 x 1536]
        gemm_wmma_kernel<<<dim3(24, 100, 1), 128, 0, stream>>>(
            z[s], qkvw[s], nullptr, qkv[s], M, 3 * Hn, Hn, Hn, 3 * Hn, 3 * Hn, 0, 0, 0, 0);
        // weight-gen: pooled query -> aw -> fused prior mask
        colmean_kernel<<<(Bn * Hn + 255) / 256, 256, 0, stream>>>(z[s], zbar[s]);
        qv_kernel<<<(Bn * Hn + 255) / 256, 256, 0, stream>>>(zbar[s], wgq, qv[s]);
        aw_kernel<<<1, 32, 0, stream>>>(qv[s], keys, awout[s]);
        priormix_kernel<<<EB, 256, 0, stream>>>(awout[s], pri[s], wm[s], BNN);
        // fused = (C @ z) @ wf + bf
        gemm_wmma_kernel<<<dim3(8, 4, Bn), 128, 0, stream>>>(
            Cm[s], z[s], nullptr, tbuf, Nn, Hn, Nn, Nn, Hn, Hn,
            (long long)Nn * Nn, (long long)Nn * Hn, (long long)Nn * Hn, 0);
        gemm_wmma_kernel<<<dim3(8, 100, 1), 128, 0, stream>>>(
            tbuf, wfuse, bfuse, fused[s], M, Hn, Hn, Hn, Hn, Hn, 0, 0, 0, 0);
        // attention (ctx written into attn buffer), then residual + fusion
        attn_kernel<<<dim3(25, NHn, Bn), 128, 0, stream>>>(qkv[s], wm[s], attn[s]);
        add3_kernel<<<EB, 256, 0, stream>>>(attn[s], x[s], fused[s], BNH);
        // FFN: xnew = attn + W2(gelu(W1 LN(attn) + b1)) + b2
        ln_kernel<<<M, 256, 0, stream>>>(attn[s], lnfg[s], lnfb[s], z[s]);  // reuse z
        gemm_wmma_kernel<<<dim3(32, 100, 1), 128, 0, stream>>>(
            z[s], fw1[s], fb1[s], h1, M, 4 * Hn, Hn, Hn, 4 * Hn, 4 * Hn, 0, 0, 0, 1);
        gemm_wmma_kernel<<<dim3(8, 100, 1), 128, 0, stream>>>(
            h1, fw2[s], fb2[s], tbuf, M, Hn, 4 * Hn, 4 * Hn, Hn, Hn, 0, 0, 0, 0);
        add2_kernel<<<EB, 256, 0, stream>>>(xnew[s], attn[s], tbuf, BNH);
        // distillation head
        gemm_wmma_kernel<<<dim3(8, 100, 1), 128, 0, stream>>>(
            xnew[s], diw[s], dib[s], proj[s], M, Hn, Hn, Hn, Hn, Hn, 0, 0, 0, 0);
        logsoftmax_kernel<<<M, 256, 0, stream>>>(proj[s], lsm[s]);
        gemm_wmma_kernel<<<dim3(8, 100, 1), 128, 0, stream>>>(
            proj[s], dow[s], dob[s], tmp2, M, Hn, Hn, Hn, Hn, Hn, 0, 0, 0, 0);
        mix_kernel<<<EB, 256, 0, stream>>>(xout[s], xnew[s], tmp2, dw, s, BNH);
    }

    // KL(batchmean) pair + distill loss
    zero_kernel<<<1, 32, 0, stream>>>(acc, 8);
    kl_kernel<<<2048, 256, 0, stream>>>(lsm[0], lsm[1], acc, BNH);
    finalize_kernel<<<1, 32, 0, stream>>>(acc, dw, loss);
}
